// DecoderContainer_39015482917297
// MI455X (gfx1250) — matrix operand order repaired and checked by reading, not compiled, not run on hardware
//
#include <hip/hip_runtime.h>
#include <hip/hip_bf16.h>

// ---------------------------------------------------------------------------
// Peak-conditioned GRU for MI455X (gfx1250, wave32, WMMA).
//   B=64, T=1024, IN=256, U=512, PEAK=512, 3U=1536
//
// v3 changes vs v2 (all on the serial scan path):
//   * pp (t-invariant) hoisted into registers before the time loop.
//   * fp32 recurrent state kept in per-wave registers (each wave owns its
//     16x16 h-patch) -> no global h reads; d_out becomes write-only and is
//     written with non-temporal stores (keeps Wr/xp hot in L2).
//   * xp slice loads issued before the WMMA loop (latency hidden behind 48
//     WMMAs); next step's xp slice prefetched (global_prefetch) pre-barrier.
//   * k-loop fully unrolled (straight-line 48-WMMA body).
// Persistent scan kernel: 16 blocks, per-step agent-scope counter barrier.
// Workspace use: ~441 MB.
// ---------------------------------------------------------------------------

typedef __attribute__((ext_vector_type(16))) __bf16 v16bf;
typedef __attribute__((ext_vector_type(8)))  float  v8f;

#define WMMA_BF16(A, Bm, Cm) \
  __builtin_amdgcn_wmma_f32_16x16x32_bf16(false, (A), false, (Bm), (short)0, (Cm), false, false)

// ---- WMMA operand loaders --------------------------------------------------
// A-matrix (16x32 bf16), bf16 row-major source: lanes 0-15 -> M=lane, kb=0;
// lanes 16-31 -> M=lane-16, kb=16. Vector elems [0..15] = K kb..kb+15
// -> two aligned 16-byte loads.
static __device__ inline v16bf load_a_bf16(const __bf16* __restrict__ A, size_t lda,
                                           int m0, int k0, int lane) {
  int m  = m0 + (lane & 15);
  int kb = (lane >> 4) << 4;                    // 0 or 16
  const __bf16* p = A + (size_t)m * lda + (size_t)(k0 + kb);
  union { uint4 u[2]; v16bf v; } t;
  t.u[0] = *(const uint4*)(p);
  t.u[1] = *(const uint4*)(p + 8);
  return t.v;
}

// B-matrix (32x16 bf16) pre-packed: blob (kt*ntiles + nt) = 32 lanes x 16 bf16
// contiguous -> two 16-byte loads per lane.
static __device__ inline v16bf load_b_packed(const __bf16* __restrict__ Bp,
                                             int ntiles, int kt, int nt, int lane) {
  const __bf16* p = Bp + (((size_t)kt * ntiles + nt) * 32 + lane) * 16;
  union { uint4 u[2]; v16bf v; } t;
  const uint4* q = (const uint4*)p;
  t.u[0] = q[0];
  t.u[1] = q[1];
  return t.v;
}

// ---- pack fp32 row-major (K x N) weights into WMMA B-operand bf16 tiles ----
__global__ void pack_b_kernel(const float* __restrict__ src, __bf16* __restrict__ dst,
                              int K, int N) {
  int idx = blockIdx.x * blockDim.x + threadIdx.x;
  int ktiles = K >> 5, ntiles = N >> 4;
  int total = ktiles * ntiles * 32;
  if (idx >= total) return;
  int lane = idx & 31;
  int tile = idx >> 5;
  int nt = tile % ntiles;
  int kt = tile / ntiles;
  int n  = (nt << 4) + (lane & 15);
  int kb = (lane >> 4) << 4;                    // lanes 0-15: K 0-15, lanes 16-31: 16-31
  int k0 = (kt << 5) + kb;
  __bf16* o = dst + (size_t)idx * 16;
#pragma unroll
  for (int j = 0; j < 8; ++j) {
    o[2*j]   = (__bf16)src[(size_t)(k0 + 2*j)     * N + n];
    o[2*j+1] = (__bf16)src[(size_t)(k0 + 2*j + 1) * N + n];
  }
}

// ---- bulk fp32 -> bf16 convert (n multiple of 4) ---------------------------
__global__ void f32_to_bf16_kernel(const float* __restrict__ src,
                                   __bf16* __restrict__ dst, int n) {
  int i = (blockIdx.x * blockDim.x + threadIdx.x) * 4;
  if (i >= n) return;
  float4 f = *(const float4*)(src + i);
  dst[i]     = (__bf16)f.x;
  dst[i + 1] = (__bf16)f.y;
  dst[i + 2] = (__bf16)f.z;
  dst[i + 3] = (__bf16)f.w;
}

// ---- generic bf16 WMMA GEMM: C(MxN) = A(bf16,MxK) * Bp + bias --------------
// One wave computes a 16x64 strip: 4 N-tiles sharing one A tile.
__global__ void gemm_bf16_kernel(const __bf16* __restrict__ A, int lda,
                                 const __bf16* __restrict__ Bp,
                                 const float* __restrict__ bias,
                                 float* __restrict__ C, int ldc,
                                 int M, int N, int K) {
  int lane = threadIdx.x & 31;
  int gw = blockIdx.x * (blockDim.x >> 5) + (threadIdx.x >> 5);
  int mtiles = M >> 4, ntiles = N >> 4, ktiles = K >> 5;
  int groups = ntiles >> 2;
  if (gw >= mtiles * groups) return;            // wave-uniform exit
  int mt = gw / groups, ng = gw % groups;
  int m0 = mt << 4, nt0 = ng << 2;
  v8f acc[4] = {{}, {}, {}, {}};
  for (int kt = 0; kt < ktiles; ++kt) {
    v16bf a  = load_a_bf16(A, (size_t)lda, m0, kt << 5, lane);
    v16bf b0 = load_b_packed(Bp, ntiles, kt, nt0 + 0, lane);
    v16bf b1 = load_b_packed(Bp, ntiles, kt, nt0 + 1, lane);
    v16bf b2 = load_b_packed(Bp, ntiles, kt, nt0 + 2, lane);
    v16bf b3 = load_b_packed(Bp, ntiles, kt, nt0 + 3, lane);
    acc[0] = WMMA_BF16(a, b0, acc[0]);
    acc[1] = WMMA_BF16(a, b1, acc[1]);
    acc[2] = WMMA_BF16(a, b2, acc[2]);
    acc[3] = WMMA_BF16(a, b3, acc[3]);
  }
  int mh = m0 + ((lane >> 4) << 3);             // lanes 16-31 hold rows m0+8..+15
#pragma unroll
  for (int j = 0; j < 4; ++j) {
    int n = ((nt0 + j) << 4) + (lane & 15);
    float bv = bias ? bias[n] : 0.0f;
#pragma unroll
    for (int r = 0; r < 8; ++r)
      C[(size_t)(mh + r) * ldc + n] = acc[j][r] + bv;
  }
}

// ---- init for the scan: zero the per-step barrier counters, h0 -> bf16 -----
__global__ void init_scan_kernel(const float* __restrict__ h0f,
                                 __bf16* __restrict__ hb0,
                                 unsigned* __restrict__ ctr) {
  int i = blockIdx.x * blockDim.x + threadIdx.x;
  if (i < 1024) ctr[i] = 0u;
  if (i < 64 * 512) hb0[i] = (__bf16)h0f[i];
}

// ---- persistent GRU scan: all 1024 steps in one kernel ---------------------
// Grid: 16 blocks x 256 threads = 128 waves.
// wave w: mt = w>>5 (batch-row tile), ut = w&31 (unit-column tile).
// Per step: hp = h_{t-1} @ Wr via 3 WMMA accumulators (z | r | hh columns),
// fused gate math; fp32 state lives in registers, bf16 state in ping-pong buf.
__global__ void gru_scan_kernel(__bf16* __restrict__ hbA,   // bf16 h, parity 0
                                __bf16* __restrict__ hbB,   // bf16 h, parity 1
                                const float* __restrict__ h0f,
                                const float* __restrict__ xp,
                                const float* __restrict__ pp,
                                const __bf16* __restrict__ Wr,
                                float* __restrict__ out,
                                unsigned* __restrict__ ctr) {
  const int U = 512, T = 1024, N3 = 1536;
  const int ntiles = N3 >> 4;                   // 96
  const unsigned NBLK = 16;
  int lane = threadIdx.x & 31;
  int w  = blockIdx.x * 8 + (threadIdx.x >> 5); // 0..127
  int mt = w >> 5;                              // 0..3
  int ut = w & 31;                              // 0..31
  int m0 = mt << 4;
  int u  = (ut << 4) + (lane & 15);
  int mh = m0 + ((lane >> 4) << 3);

  // ---- t-invariant per-lane state: pp gates + fp32 h (this wave's patch) --
  float hvr[8], pz[8], pr[8], ph[8];
#pragma unroll
  for (int r = 0; r < 8; ++r) {
    int b = mh + r;
    size_t prow = (size_t)b * N3;
    hvr[r] = h0f[(size_t)b * U + u];
    pz[r]  = pp[prow + u];
    pr[r]  = pp[prow + u + U];
    ph[r]  = pp[prow + u + 2 * U];
  }

  for (int t = 0; t < T; ++t) {
    const __bf16* hr = (t & 1) ? hbB : hbA;     // A operand (bf16)
    __bf16*       hw = (t & 1) ? hbA : hbB;

    // -- issue this step's xp loads first: latency hides behind the WMMAs --
    float xz[8], xr[8], xh[8];
#pragma unroll
    for (int r = 0; r < 8; ++r) {
      size_t xrow = ((size_t)(mh + r) * T + t) * N3;
      xz[r] = xp[xrow + u];
      xr[r] = xp[xrow + u + U];
      xh[r] = xp[xrow + u + 2 * U];
    }

    v8f accZ = {}, accR = {}, accH = {};
#pragma unroll
    for (int kt = 0; kt < 16; ++kt) {           // K = 512, fully unrolled
      v16bf a  = load_a_bf16(hr, (size_t)U, m0, kt << 5, lane);
      v16bf bz = load_b_packed(Wr, ntiles, kt, ut,      lane);
      v16bf br = load_b_packed(Wr, ntiles, kt, ut + 32, lane);
      v16bf bh = load_b_packed(Wr, ntiles, kt, ut + 64, lane);
      accZ = WMMA_BF16(a, bz, accZ);
      accR = WMMA_BF16(a, br, accR);
      accH = WMMA_BF16(a, bh, accH);
    }

#pragma unroll
    for (int r = 0; r < 8; ++r) {
      int b = mh + r;
      float z  = fminf(fmaxf(0.2f * (xz[r] + accZ[r] + pz[r]) + 0.5f, 0.0f), 1.0f);
      float rr = fminf(fmaxf(0.2f * (xr[r] + accR[r] + pr[r]) + 0.5f, 0.0f), 1.0f);
      float hh = tanhf(xh[r] + rr * (accH[r] + ph[r]));
      float hn = z * hvr[r] + (1.0f - z) * hh;
      hvr[r] = hn;                                           // fp32 chain in regs
      __builtin_nontemporal_store(hn, &out[((size_t)b * T + t) * U + u]);
      hw[(size_t)b * U + u] = (__bf16)hn;                    // bf16 for next WMMA
    }

    // -- warm L2 with next step's xp slice while we sit in the barrier --
    if (t + 1 < T) {
#pragma unroll
      for (int r = 0; r < 8; ++r) {
        size_t xrow = ((size_t)(mh + r) * T + (t + 1)) * N3;
        __builtin_prefetch(&xp[xrow + u], 0, 1);
        __builtin_prefetch(&xp[xrow + u + U], 0, 1);
        __builtin_prefetch(&xp[xrow + u + 2 * U], 0, 1);
      }
    }

    // ---- device-scope grid barrier (release fence / acquire spin) ----
    __threadfence();
    __syncthreads();
    if (threadIdx.x == 0) {
      __hip_atomic_fetch_add(&ctr[t], 1u, __ATOMIC_ACQ_REL, __HIP_MEMORY_SCOPE_AGENT);
      while (__hip_atomic_load(&ctr[t], __ATOMIC_ACQUIRE, __HIP_MEMORY_SCOPE_AGENT) < NBLK)
        __builtin_amdgcn_s_sleep(1);
    }
    __syncthreads();
    __threadfence();
  }
}

// ---------------------------------------------------------------------------
extern "C" void kernel_launch(void* const* d_in, const int* in_sizes, int n_in,
                              void* d_out, int out_size, void* d_ws, size_t ws_size,
                              hipStream_t stream) {
  const int B = 64, T = 1024, IN = 256, U = 512, PD = 512, N3 = 3 * U;

  const float* x     = (const float*)d_in[0];   // (B,T,IN)
  const float* peak  = (const float*)d_in[1];   // (B,PD)
  const float* Wk_f  = (const float*)d_in[2];   // (IN,3U)
  const float* Wr_f  = (const float*)d_in[3];   // (U,3U)
  const float* Wp_f  = (const float*)d_in[4];   // (PD,3U)
  const float* bias  = (const float*)d_in[5];   // (3U)
  const float* Wi_f  = (const float*)d_in[6];   // (PD,U)
  float* out = (float*)d_out;                   // (B,T,U)

  // ---- workspace carve-up (~441 MB total) ----
  char* ws = (char*)d_ws;
  size_t off = 0;
  auto alloc = [&](size_t bytes) -> void* {
    void* p = ws + off;
    off += (bytes + 255) & ~(size_t)255;
    return p;
  };
  __bf16*   Wk   = (__bf16*)alloc((size_t)IN * N3 * 2);      // packed
  __bf16*   Wr   = (__bf16*)alloc((size_t)U  * N3 * 2);      // packed
  __bf16*   Wp   = (__bf16*)alloc((size_t)PD * N3 * 2);      // packed
  __bf16*   Wi   = (__bf16*)alloc((size_t)PD * U  * 2);      // packed
  __bf16*   x_bf = (__bf16*)alloc((size_t)B * T * IN * 2);
  __bf16*   p_bf = (__bf16*)alloc((size_t)B * PD * 2);
  __bf16*   hbA  = (__bf16*)alloc((size_t)B * U * 2);
  __bf16*   hbB  = (__bf16*)alloc((size_t)B * U * 2);
  float*    h0f  = (float*)alloc((size_t)B * U * 4);
  float*    pp   = (float*)alloc((size_t)B * N3 * 4);
  unsigned* ctr  = (unsigned*)alloc((size_t)T * 4);
  float*    xp   = (float*)alloc((size_t)B * T * N3 * 4);    // 403 MB

  // ---- pack weights into WMMA B-operand layout ----
  {
    int tot;
    tot = (IN >> 5) * (N3 >> 4) * 32;
    pack_b_kernel<<<(tot + 255) / 256, 256, 0, stream>>>(Wk_f, Wk, IN, N3);
    tot = (U >> 5) * (N3 >> 4) * 32;
    pack_b_kernel<<<(tot + 255) / 256, 256, 0, stream>>>(Wr_f, Wr, U, N3);
    tot = (PD >> 5) * (N3 >> 4) * 32;
    pack_b_kernel<<<(tot + 255) / 256, 256, 0, stream>>>(Wp_f, Wp, PD, N3);
    tot = (PD >> 5) * (U >> 4) * 32;
    pack_b_kernel<<<(tot + 255) / 256, 256, 0, stream>>>(Wi_f, Wi, PD, U);
  }

  // ---- bf16 copies of the A-side activations ----
  {
    int n = B * T * IN;                          // 16.7M
    f32_to_bf16_kernel<<<(n / 4 + 255) / 256, 256, 0, stream>>>(x, x_bf, n);
    n = B * PD;
    f32_to_bf16_kernel<<<(n / 4 + 255) / 256, 256, 0, stream>>>(peak, p_bf, n);
  }

  // ---- h0 = peak @ peak_initial ----
  {
    int waves = (B >> 4) * ((U >> 4) >> 2);      // 32
    gemm_bf16_kernel<<<(waves * 32 + 255) / 256, 256, 0, stream>>>(
        p_bf, PD, Wi, nullptr, h0f, U, B, U, PD);
  }
  // ---- pp = peak @ peak_kernel ----
  {
    int waves = (B >> 4) * ((N3 >> 4) >> 2);     // 96
    gemm_bf16_kernel<<<(waves * 32 + 255) / 256, 256, 0, stream>>>(
        p_bf, PD, Wp, nullptr, pp, N3, B, N3, PD);
  }
  // ---- xp = x @ kernel + bias ----
  {
    int M = B * T;
    int waves = (M >> 4) * ((N3 >> 4) >> 2);     // 98304
    gemm_bf16_kernel<<<(waves * 32 + 255) / 256, 256, 0, stream>>>(
        x_bf, IN, Wk, bias, xp, N3, M, N3, IN);
  }

  // ---- scan init: zero barrier counters, h0 -> bf16 ----
  init_scan_kernel<<<(B * U + 255) / 256, 256, 0, stream>>>(h0f, hbA, ctr);

  // ---- persistent sequential scan: one kernel, 1024 internal steps ----
  gru_scan_kernel<<<16, 256, 0, stream>>>(hbA, hbB, h0f, xp, pp, Wr, out, ctr);
}
